// QuantizedLinear_15865609192074
// MI455X (gfx1250) — matrix operand order repaired and checked by reading, not compiled, not run on hardware
//
#include <hip/hip_runtime.h>
#include <hip/hip_bf16.h>

typedef __attribute__((ext_vector_type(16))) _Float16 v16h;
typedef __attribute__((ext_vector_type(8)))  _Float16 v8h;
typedef __attribute__((ext_vector_type(4)))  _Float16 v4h;
typedef __attribute__((ext_vector_type(8)))  float    v8f;

// Problem constants (from reference setup_inputs)
constexpr int M_DIM = 4 * 2048;   // B*S = 8192
constexpr int N_DIM = 11008;      // D_OUT
constexpr int K_DIM = 4096;       // D_IN
constexpr int NGROUPS = 32;       // K / 128

// Tiling: 128(M) x 256(N) per workgroup, 8 waves as 2x4, each wave 64x64.
// BK=32 so double-buffered LDS fits: 2*(128+256)*40*2 = 60 KB.
constexpr int BM = 128;
constexpr int BN = 256;
constexpr int BK = 32;
constexpr int LDSK = BK + 8;      // padded row stride in halfs (80 B, 16B-aligned)

__global__ __launch_bounds__(256)
void QuantizedLinear_15865609192074_kernel(const float* __restrict__ x,
                                           const float* __restrict__ scales,
                                           const float* __restrict__ bias,
                                           const int*   __restrict__ qw,
                                           const int*   __restrict__ zeros,
                                           float*       __restrict__ y)
{
    __shared__ _Float16 xs[2][BM * LDSK];   // x tiles, f16   (2 x 10 KB)
    __shared__ _Float16 ws[2][BN * LDSK];   // W tiles, f16   (2 x 20 KB)

    const int tid     = threadIdx.x;
    const int lane    = tid & 31;
    const int wave    = tid >> 5;        // 0..7
    const int wm      = wave & 1;        // M sub-tile (64 rows)
    const int wn      = wave >> 1;       // N sub-tile (64 cols)
    const int l16     = lane & 15;
    const int halfSel = lane >> 4;       // 0: lanes 0-15, 1: lanes 16-31

    const int bm = blockIdx.y * BM;
    const int bn = blockIdx.x * BN;

    v8f acc[4][4] = {};

    // ---- pipeline staging registers ----
    float4 xv[4];
    int4   qv[8];
    float  sv[8], zv[8];

    // Issue global loads for K-tile at k0 into registers.
    auto load_tile = [&](int k0) {
        const int g = k0 >> 7;           // quant group (128 wide)
        #pragma unroll
        for (int i = 0; i < 4; ++i) {    // x: 128 rows x 8 float4-chunks
            const int v   = tid + i * 256;
            const int row = v >> 3;
            const int c4  = (v & 7) << 2;
            xv[i] = *(const float4*)(x + (size_t)(bm + row) * K_DIM + k0 + c4);
        }
        #pragma unroll
        for (int i = 0; i < 8; ++i) {    // qw: 256 rows x 8 int4-chunks
            const int v   = tid + i * 256;
            const int row = v >> 3;
            const int c4  = (v & 7) << 2;
            const int n   = bn + row;
            qv[i] = *(const int4*)(qw + (size_t)n * K_DIM + k0 + c4);
            sv[i] = scales[n * NGROUPS + g];
            zv[i] = (float)zeros[n * NGROUPS + g];
        }
    };

    // Convert/dequantize staged registers into LDS buffer `buf`.
    auto store_tile = [&](int buf) {
        #pragma unroll
        for (int i = 0; i < 4; ++i) {
            const int v   = tid + i * 256;
            const int row = v >> 3;
            const int c4  = (v & 7) << 2;
            v4h h;
            h[0] = (_Float16)xv[i].x;
            h[1] = (_Float16)xv[i].y;
            h[2] = (_Float16)xv[i].z;
            h[3] = (_Float16)xv[i].w;
            *(v4h*)&xs[buf][row * LDSK + c4] = h;
        }
        #pragma unroll
        for (int i = 0; i < 8; ++i) {
            const int v   = tid + i * 256;
            const int row = v >> 3;
            const int c4  = (v & 7) << 2;
            v4h h;
            h[0] = (_Float16)(sv[i] * ((float)qv[i].x - zv[i]));
            h[1] = (_Float16)(sv[i] * ((float)qv[i].y - zv[i]));
            h[2] = (_Float16)(sv[i] * ((float)qv[i].z - zv[i]));
            h[3] = (_Float16)(sv[i] * ((float)qv[i].w - zv[i]));
            *(v4h*)&ws[buf][row * LDSK + c4] = h;
        }
    };

    // ---- prologue: stage tile 0 ----
    load_tile(0);
    store_tile(0);
    __syncthreads();

    for (int k0 = 0; k0 < K_DIM; k0 += BK) {
        const int cur  = (k0 / BK) & 1;
        const bool more = (k0 + BK) < K_DIM;

        // Issue next tile's global loads; compiler defers s_wait_loadcnt until
        // store_tile below, so they overlap the whole WMMA block.
        if (more) load_tile(k0 + BK);

        // Warm L2 two tiles ahead.
        if (k0 + 2 * BK < K_DIM) {
            __builtin_prefetch(x  + (size_t)(bm + (tid >> 1)) * K_DIM + k0 + 2 * BK, 0, 1);
            __builtin_prefetch(qw + (size_t)(bn + tid) * K_DIM + k0 + 2 * BK, 0, 1);
        }

        // ---- compute from buffer `cur`: 8 fragment loads, 16 WMMAs ----
        {
            v16h bf[4];
            #pragma unroll
            for (int ni = 0; ni < 4; ++ni) {
                // B: lane l16 = N column; K pattern 0-7/16-23 (lanes<16), 8-15/24-31 (lanes>=16)
                const _Float16* bp = &ws[cur][(wn * 64 + ni * 16 + l16) * LDSK + halfSel * 8];
                union { v16h v; v8h h[2]; } u;
                u.h[0] = *(const v8h*)bp;
                u.h[1] = *(const v8h*)(bp + 16);
                bf[ni] = u.v;
            }
            v16h af[4];
            #pragma unroll
            for (int mi = 0; mi < 4; ++mi) {
                // A: lane l16 = M row; K pattern 0-7/16-23 (lanes<16), 8-15/24-31 (lanes>=16)
                const _Float16* ap = &xs[cur][(wm * 64 + mi * 16 + l16) * LDSK + halfSel * 8];
                union { v16h v; v8h h[2]; } u;
                u.h[0] = *(const v8h*)ap;
                u.h[1] = *(const v8h*)(ap + 16);
                af[mi] = u.v;
            }
            #pragma unroll
            for (int mi = 0; mi < 4; ++mi) {
                #pragma unroll
                for (int ni = 0; ni < 4; ++ni) {
                    acc[mi][ni] = __builtin_amdgcn_wmma_f32_16x16x32_f16(
                        false, af[mi], false, bf[ni],
                        (short)0, acc[mi][ni], false, false);
                }
            }
        }

        // ---- stage next tile into the other buffer; single barrier per iter ----
        if (more) store_tile(cur ^ 1);
        __syncthreads();
    }

    // ---- epilogue: bias add + f32 store ----
    // C/D layout: VGPR element e -> row (e + 8*halfSel), lane l16 -> col
    #pragma unroll
    for (int ni = 0; ni < 4; ++ni) {
        const int col = bn + wn * 64 + ni * 16 + l16;
        const float bv = bias[col];
        #pragma unroll
        for (int mi = 0; mi < 4; ++mi) {
            #pragma unroll
            for (int e = 0; e < 8; ++e) {
                const int row = bm + wm * 64 + mi * 16 + halfSel * 8 + e;
                y[(size_t)row * N_DIM + col] = acc[mi][ni][e] + bv;
            }
        }
    }
}

extern "C" void kernel_launch(void* const* d_in, const int* in_sizes, int n_in,
                              void* d_out, int out_size, void* d_ws, size_t ws_size,
                              hipStream_t stream) {
    (void)in_sizes; (void)n_in; (void)out_size; (void)d_ws; (void)ws_size;
    const float* x      = (const float*)d_in[0];
    const float* scales = (const float*)d_in[1];
    const float* bias   = (const float*)d_in[2];
    const int*   qw     = (const int*)d_in[3];
    const int*   zeros  = (const int*)d_in[4];
    float* y = (float*)d_out;

    dim3 grid(N_DIM / BN, M_DIM / BM);   // 43 x 64 workgroups
    QuantizedLinear_15865609192074_kernel<<<grid, 256, 0, stream>>>(x, scales, bias, qw, zeros, y);
}
